// MultiHeadAttention_40252433498195
// MI455X (gfx1250) — matrix operand order repaired and checked by reading, not compiled, not run on hardware
//
#include <hip/hip_runtime.h>

// ---------------------------------------------------------------------------
// Problem constants (reference: B=4, N=2048, E=512, H=8, D=64)
// ---------------------------------------------------------------------------
constexpr int Bq  = 4;
constexpr int SEQ = 2048;
constexpr int EMB = 512;
constexpr int NH  = 8;
constexpr int HD  = 64;
constexpr int MTOT = Bq * SEQ;

typedef __attribute__((ext_vector_type(16))) __bf16        v16bf;
typedef __attribute__((ext_vector_type(8)))  float         v8f;
typedef __attribute__((ext_vector_type(8)))  unsigned int  v8u;
typedef __attribute__((ext_vector_type(4)))  int           v4i;
typedef __attribute__((ext_vector_type(2)))  _Float16      h2;

// ---------------------------------------------------------------------------
// gfx1250 async global->LDS DMA (present on this toolchain; guarded fallback)
// ---------------------------------------------------------------------------
#if __has_builtin(__builtin_amdgcn_global_load_async_to_lds_b128) && \
    __has_builtin(__builtin_amdgcn_s_wait_asynccnt)
#define HAVE_ASYNC_LDS 1
typedef __attribute__((address_space(1))) v4i* gv4i_p;
typedef __attribute__((address_space(3))) v4i* lv4i_p;
__device__ __forceinline__ void cp16_async(const void* g, void* l) {
    __builtin_amdgcn_global_load_async_to_lds_b128((gv4i_p)g, (lv4i_p)l, 0, 0);
}
__device__ __forceinline__ void wait_async() {
    __builtin_amdgcn_s_wait_asynccnt(0);
}
#else
#define HAVE_ASYNC_LDS 0
__device__ __forceinline__ void cp16_async(const void* g, void* l) {
    *(uint4*)l = *(const uint4*)g;
}
__device__ __forceinline__ void wait_async() {}
#endif

// f32 -> bf16, round to nearest even
__device__ __forceinline__ unsigned short f2bf(float f) {
    unsigned int u = __builtin_bit_cast(unsigned int, f);
    u += 0x7FFFu + ((u >> 16) & 1u);
    return (unsigned short)(u >> 16);
}
__device__ __forceinline__ unsigned int pack_bf2(float lo, float hi) {
    return (unsigned int)f2bf(lo) | ((unsigned int)f2bf(hi) << 16);
}

__device__ __forceinline__ v8f zero8() {
    v8f z = {0.f, 0.f, 0.f, 0.f, 0.f, 0.f, 0.f, 0.f};
    return z;
}

// A-operand fragment (16x32 bf16, MxK), CDNA5 16-bit A layout.
__device__ __forceinline__ v16bf load_a(const unsigned short* base, int stride) {
    const int lane = threadIdx.x & 31;
    const int r = lane & 15, g = lane >> 4;
    v8u u;
#pragma unroll
    for (int i = 0; i < 8; ++i) {
        const int k = ((i < 4) ? 0 : 16) + g * 8 + ((i & 3) << 1);
        u[i] = *(const unsigned int*)(base + (size_t)r * stride + k);
    }
    return __builtin_bit_cast(v16bf, u);
}

// B-operand fragment (32x16 bf16, KxN) from row-major [N][K] ("Bt") tile.
__device__ __forceinline__ v16bf load_b(const unsigned short* base, int stride) {
    const int lane = threadIdx.x & 31;
    const int n = lane & 15, g = lane >> 4;
    v8u u;
#pragma unroll
    for (int i = 0; i < 8; ++i) {
        const int k = ((i < 4) ? 0 : 16) + g * 8 + ((i & 3) << 1);
        u[i] = *(const unsigned int*)(base + (size_t)n * stride + k);
    }
    return __builtin_bit_cast(v16bf, u);
}

__device__ __forceinline__ v8f wmma_bf16(v16bf a, v16bf b, v8f c) {
    return __builtin_amdgcn_wmma_f32_16x16x32_bf16(false, a, false, b,
                                                   (short)0, c, false, false);
}

// ---------------------------------------------------------------------------
// GEMM: OUT = X[M][K=EMB] * W[N=EMB][K=EMB]^T, double-buffered LDS pipeline.
// MODE 0: bf16 [B,H,SEQ,HD]             (K projection)
// MODE 3: bf16 [B,H,SEQ,HD] * 1/sqrt(E) (Q projection, scale folded in)
// MODE 1: bf16 [B,H,HD,SEQ]             (V projection, transposed)
// MODE 2: f32  [M][EMB]                 (output projection)
// ---------------------------------------------------------------------------
template <int MODE>
__global__ void __launch_bounds__(256)
gemm_xwt(const float* __restrict__ X, const float* __restrict__ W,
         void* __restrict__ outp) {
    __shared__ unsigned short lsA[2][64][40];
    __shared__ unsigned short lsB[2][128][40];

    const int tid  = threadIdx.x;
    const int lane = tid & 31;
    const int wave = tid >> 5;
    const int wm = wave >> 1;
    const int wn = wave & 1;
    const int m0 = blockIdx.y * 64;
    const int n0 = blockIdx.x * 128;

    float4 ra[2], rb[4];

    auto fetch = [&](int k0) {
#pragma unroll
        for (int it = 0; it < 2; ++it) {
            const int idx = tid + it * 256;
            ra[it] = *(const float4*)(X + (size_t)(m0 + (idx >> 3)) * EMB + k0 + (idx & 7) * 4);
        }
#pragma unroll
        for (int it = 0; it < 4; ++it) {
            const int idx = tid + it * 256;
            rb[it] = *(const float4*)(W + (size_t)(n0 + (idx >> 3)) * EMB + k0 + (idx & 7) * 4);
        }
    };
    auto stage = [&](int buf) {
#pragma unroll
        for (int it = 0; it < 2; ++it) {
            const int idx = tid + it * 256;
            uint2 p;
            p.x = pack_bf2(ra[it].x, ra[it].y);
            p.y = pack_bf2(ra[it].z, ra[it].w);
            *(uint2*)(&lsA[buf][idx >> 3][(idx & 7) * 4]) = p;
        }
#pragma unroll
        for (int it = 0; it < 4; ++it) {
            const int idx = tid + it * 256;
            uint2 p;
            p.x = pack_bf2(rb[it].x, rb[it].y);
            p.y = pack_bf2(rb[it].z, rb[it].w);
            *(uint2*)(&lsB[buf][idx >> 3][(idx & 7) * 4]) = p;
        }
    };

    v8f acc[4];
#pragma unroll
    for (int t = 0; t < 4; ++t) acc[t] = zero8();

    constexpr int KT = EMB / 32;   // 16 K-steps
    fetch(0);
    stage(0);
    __syncthreads();

    for (int kt = 0; kt < KT; ++kt) {
        const int cur = kt & 1;
        if (kt + 1 < KT) fetch((kt + 1) * 32);   // global loads fly over WMMA

        const v16bf af = load_a(&lsA[cur][wm * 16][0], 40);
        v16bf bfr[4];
#pragma unroll
        for (int t = 0; t < 4; ++t)
            bfr[t] = load_b(&lsB[cur][wn * 64 + t * 16][0], 40);
#pragma unroll
        for (int t = 0; t < 4; ++t)
            acc[t] = wmma_bf16(af, bfr[t], acc[t]);

        if (kt + 1 < KT) stage(cur ^ 1);         // fill the other buffer
        __syncthreads();
    }

    // Epilogue. C layout: VGPR j -> row (j + 8*(lane>>4)), col (lane&15).
    const int col = lane & 15, g = lane >> 4;
#pragma unroll
    for (int t = 0; t < 4; ++t) {
#pragma unroll
        for (int j = 0; j < 8; ++j) {
            const int m = m0 + wm * 16 + j + 8 * g;
            const int n = n0 + wn * 64 + t * 16 + col;
            float v = acc[t][j];
            if constexpr (MODE == 2) {
                ((float*)outp)[(size_t)m * EMB + n] = v;
            } else {
                if constexpr (MODE == 3) v *= 0.044194173824159216f; // 1/sqrt(512)
                const int bb = m >> 11, s = m & (SEQ - 1);   // SEQ = 2048
                const int hh = n >> 6,  d = n & (HD - 1);    // HD  = 64
                const size_t idx = (MODE == 1)
                    ? ((((size_t)bb * NH + hh) * HD + d) * SEQ + s)
                    : ((((size_t)bb * NH + hh) * SEQ + s) * HD + d);
                ((unsigned short*)outp)[idx] = f2bf(v);
            }
        }
    }
}

// ---------------------------------------------------------------------------
// Flash attention: grid (SEQ/128, B*H), 8 waves/block, 16 query rows/wave.
// Double-buffered async-DMA K/V streaming, 32 keys per iteration.
// Row-sums via WMMA-with-ones; row-max via packed-f16 butterflies.
// ---------------------------------------------------------------------------
__global__ void __launch_bounds__(256)
attn_kernel(const unsigned short* __restrict__ qb,
            const unsigned short* __restrict__ kb,
            const unsigned short* __restrict__ vtb,
            float* __restrict__ attnout) {
    __shared__ unsigned short lsK[2][32][72];
    __shared__ unsigned short lsV[2][64][40];
    __shared__ unsigned short lsP[8][16][40];

    const int bh = blockIdx.y;
    const int b = bh / NH, h = bh % NH;
    const unsigned short* qh = qb  + (size_t)bh * SEQ * HD;
    const unsigned short* kh = kb  + (size_t)bh * SEQ * HD;
    const unsigned short* vh = vtb + (size_t)bh * HD * SEQ;

    const int tid  = threadIdx.x;
    const int lane = tid & 31;
    const int wave = tid >> 5;
    const int qrow0 = blockIdx.x * 128 + wave * 16;

    v16bf qf[2];   // Q is pre-scaled by 1/sqrt(E) at projection time
#pragma unroll
    for (int s = 0; s < 2; ++s)
        qf[s] = load_a(qh + (size_t)qrow0 * HD + s * 32, HD);

    // all-ones B fragment: rowsum(P) = P x Ones, replicated across columns
    v8u ou;
#pragma unroll
    for (int i = 0; i < 8; ++i) ou[i] = 0x3F803F80u;   // bf16 1.0 pair
    const v16bf onesb = __builtin_bit_cast(v16bf, ou);

    float mrow[8], lrow[8];
#pragma unroll
    for (int j = 0; j < 8; ++j) { mrow[j] = -3.0e38f; lrow[j] = 0.f; }
    v8f o[4];
#pragma unroll
    for (int t = 0; t < 4; ++t) o[t] = zero8();

    // per-thread staging coordinates (one 16B chunk each, both tiles)
    const int kr = tid >> 3, kc = tid & 7;   // K tile: 32 rows x 8 chunks
    const int vr = tid >> 2, vc = tid & 3;   // V tile: 64 rows x 4 chunks

    auto issue_copy = [&](int nk0, int buf) {
        cp16_async(kh + (size_t)(nk0 + kr) * HD + kc * 8, &lsK[buf][kr][kc * 8]);
        cp16_async(vh + (size_t)vr * SEQ + nk0 + vc * 8,  &lsV[buf][vr][vc * 8]);
    };

    constexpr int NIT = SEQ / 32;    // 64 key blocks
    issue_copy(0, 0);
    wait_async();
    __syncthreads();

    for (int it = 0; it < NIT; ++it) {
        const int cur = it & 1;
        const int nk0 = it * 32;
        if (it + 1 < NIT) issue_copy(nk0 + 32, cur ^ 1);  // DMA next block now

        // S = Q * K^T: preload all 4 B fragments, then 4 WMMAs
        v16bf bk00 = load_b(&lsK[cur][0][0],   72);
        v16bf bk10 = load_b(&lsK[cur][16][0],  72);
        v16bf bk01 = load_b(&lsK[cur][0][32],  72);
        v16bf bk11 = load_b(&lsK[cur][16][32], 72);
        v8f s0 = zero8(), s1 = zero8();
        s0 = wmma_bf16(qf[0], bk00, s0);
        s1 = wmma_bf16(qf[0], bk10, s1);
        s0 = wmma_bf16(qf[1], bk01, s0);
        s1 = wmma_bf16(qf[1], bk11, s1);

        // row-max: convert both tiles to packed f16, single pk-max per pair,
        // then packed butterflies over 16 lanes (all in v_pk_max_num_f16)
        h2 hp[4];
#pragma unroll
        for (int jj = 0; jj < 4; ++jj) {
            h2 x, y;
            x.x = (_Float16)s0[2 * jj];
            x.y = (_Float16)s0[2 * jj + 1];
            y.x = (_Float16)s1[2 * jj];
            y.y = (_Float16)s1[2 * jj + 1];
            hp[jj] = __builtin_elementwise_max(x, y);
        }
#pragma unroll
        for (int msk = 1; msk <= 8; msk <<= 1) {
#pragma unroll
            for (int jj = 0; jj < 4; ++jj) {
                const int u = __shfl_xor(__builtin_bit_cast(int, hp[jj]), msk, 32);
                hp[jj] = __builtin_elementwise_max(hp[jj],
                                                   __builtin_bit_cast(h2, u));
            }
        }

        float alpha[8];
#pragma unroll
        for (int j = 0; j < 8; ++j) {
            const float mx = (j & 1) ? (float)hp[j >> 1].y : (float)hp[j >> 1].x;
            const float mnew = fmaxf(mrow[j], mx);
            alpha[j] = __expf(mrow[j] - mnew);
            s0[j] = __expf(s0[j] - mnew);
            s1[j] = __expf(s1[j] - mnew);
            mrow[j] = mnew;
        }
#pragma unroll
        for (int t = 0; t < 4; ++t)
#pragma unroll
            for (int j = 0; j < 8; ++j) o[t][j] *= alpha[j];

        // C-layout -> A-layout via per-wave LDS round trip
        unsigned short* pbase = &lsP[wave][0][0];
        const int col = lane & 15, g = lane >> 4;
#pragma unroll
        for (int j = 0; j < 8; ++j) {
            const int row = j + 8 * g;
            pbase[row * 40 + col]      = f2bf(s0[j]);
            pbase[row * 40 + 16 + col] = f2bf(s1[j]);
        }
        const v16bf pa = load_a(pbase, 40);

        // row sums with one extra WMMA (replicated across columns)
        const v8f srow = wmma_bf16(pa, onesb, zero8());

        v16bf bv[4];
#pragma unroll
        for (int t = 0; t < 4; ++t)
            bv[t] = load_b(&lsV[cur][t * 16][0], 40);
#pragma unroll
        for (int t = 0; t < 4; ++t)
            o[t] = wmma_bf16(pa, bv[t], o[t]);

#pragma unroll
        for (int j = 0; j < 8; ++j)
            lrow[j] = lrow[j] * alpha[j] + srow[j];

        wait_async();      // next block's DMA must land before the barrier
        __syncthreads();
    }

    // normalize and write f32 [B, SEQ, EMB] (heads merged)
    const int col = lane & 15, g = lane >> 4;
    float* outb = attnout + (size_t)b * SEQ * EMB + (size_t)h * HD;
#pragma unroll
    for (int j = 0; j < 8; ++j) {
        const float rl = 1.0f / lrow[j];
        const int m = qrow0 + j + 8 * g;
#pragma unroll
        for (int t = 0; t < 4; ++t)
            outb[(size_t)m * EMB + t * 16 + col] = o[t][j] * rl;
    }
}

// ---------------------------------------------------------------------------
// Host-side orchestration
// ---------------------------------------------------------------------------
extern "C" void kernel_launch(void* const* d_in, const int* in_sizes, int n_in,
                              void* d_out, int out_size, void* d_ws, size_t ws_size,
                              hipStream_t stream) {
    (void)in_sizes; (void)n_in; (void)out_size; (void)ws_size;
    const float* query = (const float*)d_in[0];
    const float* key_  = (const float*)d_in[1];
    const float* value = (const float*)d_in[2];
    const float* Wq    = (const float*)d_in[3];
    const float* Wk    = (const float*)d_in[4];
    const float* Wv    = (const float*)d_in[5];
    const float* Wo    = (const float*)d_in[6];

    const size_t nQKV = (size_t)Bq * NH * SEQ * HD;
    unsigned short* qbuf = (unsigned short*)d_ws;
    unsigned short* kbuf = qbuf + nQKV;
    unsigned short* vtb  = kbuf + nQKV;
    float* attnout = (float*)(vtb + nQKV);

    dim3 ggrid(EMB / 128, MTOT / 64);
    gemm_xwt<3><<<ggrid, 256, 0, stream>>>(query, Wq, (void*)qbuf);  // Q (scaled)
    gemm_xwt<0><<<ggrid, 256, 0, stream>>>(key_,  Wk, (void*)kbuf);  // K
    gemm_xwt<1><<<ggrid, 256, 0, stream>>>(value, Wv, (void*)vtb);   // V^T

    attn_kernel<<<dim3(SEQ / 128, Bq * NH), 256, 0, stream>>>(qbuf, kbuf, vtb, attnout);

    gemm_xwt<2><<<ggrid, 256, 0, stream>>>(attnout, Wo, d_out);
}
